// MarketGuidedAttention_7765300871546
// MI455X (gfx1250) — matrix operand order repaired and checked by reading, not compiled, not run on hardware
//
#include <hip/hip_runtime.h>
#include <hip/hip_bf16.h>

// Problem constants
#define B_  2
#define L_  2048
#define D_  1024
#define H_  16
#define DH_ 64
#define BL_ (B_ * L_)     // 4096
#define D3_ (3 * D_)      // 3072

typedef __attribute__((ext_vector_type(16))) __bf16 v16bf;
typedef __attribute__((ext_vector_type(8)))  float  v8f;

#define WMMA_BF16(a, b, c) \
  __builtin_amdgcn_wmma_f32_16x16x32_bf16(false, (a), false, (b), (short)0, (c), false, false)

union V16U {
  v16bf v;
  float4 q[2];
};

// B-fragment (32x16, bf16): lane n=l&15 reads 16 contiguous bf16 (K 0..15 lo / 16..31 hi)
__device__ __forceinline__ v16bf load_bfrag(const __bf16* p) {
  return *reinterpret_cast<const v16bf*>(p);
}

// A-fragment (16x32, bf16): lane m=l&15 reads 16 contiguous bf16 {koff..koff+15}
__device__ __forceinline__ v16bf load_afrag(const __bf16* base, int akoff) {
  V16U u;
  u.q[0] = *reinterpret_cast<const float4*>(base + akoff);
  u.q[1] = *reinterpret_cast<const float4*>(base + 8 + akoff);
  return u.v;
}

// ---------------------------------------------------------------------------
// Kernel 1: fp32 -> bf16 conversion (grid-stride)
// ---------------------------------------------------------------------------
__global__ void k_f32_to_bf16(const float* __restrict__ s, __bf16* __restrict__ d, int n) {
  int i = blockIdx.x * blockDim.x + threadIdx.x;
  int stride = gridDim.x * blockDim.x;
  for (; i < n; i += stride) d[i] = (__bf16)s[i];
}

// ---------------------------------------------------------------------------
// Kernel 2: QKV projection. qkv = x @ W_in^T + b_in, routed per-head:
//   Q[bh][l][d], K[bh][l][d] row-major; V stored transposed Vt[bh][d][l].
// One wave computes a 16(M) x 64(N) tile; 8 waves / block.
// ---------------------------------------------------------------------------
__global__ __launch_bounds__(256, 1)
void k_qkv(const __bf16* __restrict__ xbf, const __bf16* __restrict__ wbf,
           const float* __restrict__ bias,
           __bf16* __restrict__ Qb, __bf16* __restrict__ Kb, __bf16* __restrict__ Vt) {
  const int lane  = threadIdx.x & 31;
  const int wavei = threadIdx.x >> 5;
  const int job   = blockIdx.x * 8 + wavei;       // 12288 jobs = 256 mtiles * 48 ntiles
  const int mt    = job / 48;
  const int n064  = (job % 48) * 64;
  const int m0    = mt * 16;
  const int col   = lane & 15;
  const int half  = lane >> 4;
  const int akoff = half ? 16 : 0;
  const int bkoff = half ? 16 : 0;

  v8f zero = {0.f, 0.f, 0.f, 0.f, 0.f, 0.f, 0.f, 0.f};
  v8f c[4];
#pragma unroll
  for (int nt = 0; nt < 4; ++nt) c[nt] = zero;

  const __bf16* arow = xbf + (size_t)(m0 + col) * D_;
  for (int k0 = 0; k0 < D_; k0 += 32) {
    v16bf a = load_afrag(arow + k0, akoff);
#pragma unroll
    for (int nt = 0; nt < 4; ++nt) {
      const __bf16* brow = wbf + (size_t)(n064 + nt * 16 + col) * D_ + k0 + bkoff;
      c[nt] = WMMA_BF16(a, load_bfrag(brow), c[nt]);
    }
  }

  const int region = n064 >> 10;          // 0=Q, 1=K, 2=V
  const int h      = (n064 & 1023) >> 6;  // head
  const int b      = m0 >> 11;
  const int l0     = m0 & 2047;
  const size_t bh  = (size_t)(b * H_ + h);

#pragma unroll
  for (int nt = 0; nt < 4; ++nt) {
    const int e  = n064 + nt * 16 + col;
    const float bv = bias[e];
    const int d = e & 63;
#pragma unroll
    for (int r = 0; r < 8; ++r) {
      const int lq = l0 + r + half * 8;
      __bf16 v = (__bf16)(c[nt][r] + bv);
      if (region == 0)      Qb[(bh * L_ + lq) * DH_ + d] = v;
      else if (region == 1) Kb[(bh * L_ + lq) * DH_ + d] = v;
      else                  Vt[(bh * DH_ + d) * L_ + lq] = v;
    }
  }
}

// ---------------------------------------------------------------------------
// Kernel 3: flash attention per (b,h). Each wave owns 16 Q rows; online
// softmax over key chunks of 32. P relayout C-frag -> A-frag via per-wave LDS.
// ---------------------------------------------------------------------------
__global__ __launch_bounds__(128, 1)
void k_flash(const __bf16* __restrict__ Qb, const __bf16* __restrict__ Kb,
             const __bf16* __restrict__ Vt, const float* __restrict__ market,
             __bf16* __restrict__ ctx) {
  __shared__ __attribute__((aligned(32))) __bf16 pbuf[4][16 * 32];

  const int lane  = threadIdx.x & 31;
  const int w     = threadIdx.x >> 5;
  const int bh    = blockIdx.x >> 5;       // 32 (b,h) pairs
  const int mblk  = blockIdx.x & 31;       // 32 m-blocks of 64 rows
  const int b     = bh >> 4;
  const int h     = bh & 15;
  const int m0    = mblk * 64 + w * 16;
  const int col   = lane & 15;
  const int half  = lane >> 4;
  const int akoff = half ? 16 : 0;
  const int bkoff = half ? 16 : 0;
  const float vol   = market[b * 8];       // market_info[b,0]
  const float scale = 0.125f;              // 1/sqrt(64)

  const __bf16* qrow = Qb + ((size_t)bh * L_ + m0 + col) * DH_;
  v16bf aq0 = load_afrag(qrow, akoff);        // dh 0..31
  v16bf aq1 = load_afrag(qrow + 32, akoff);   // dh 32..63

  v8f zero = {0.f, 0.f, 0.f, 0.f, 0.f, 0.f, 0.f, 0.f};
  v8f o0 = zero, o1 = zero, o2 = zero, o3 = zero;
  float mrun[8], lrun[8];
#pragma unroll
  for (int r = 0; r < 8; ++r) { mrun[r] = -1e30f; lrun[r] = 0.f; }

  __bf16* pw = &pbuf[w][0];

  for (int j0 = 0; j0 < L_; j0 += 32) {
    // --- scores S (16 x 32), fp32 accum over DH=64 (2 k-steps) ---
    const __bf16* krow0 = Kb + ((size_t)bh * L_ + j0 + col) * DH_ + bkoff;
    const __bf16* krow1 = Kb + ((size_t)bh * L_ + j0 + 16 + col) * DH_ + bkoff;
    v8f s0 = zero, s1 = zero;
    s0 = WMMA_BF16(aq0, load_bfrag(krow0), s0);
    s0 = WMMA_BF16(aq1, load_bfrag(krow0 + 32), s0);
    s1 = WMMA_BF16(aq0, load_bfrag(krow1), s1);
    s1 = WMMA_BF16(aq1, load_bfrag(krow1 + 32), s1);

    // --- online softmax (row stats live per half-wave: row r / r+8) ---
    float mnew[8], alpha[8];
#pragma unroll
    for (int r = 0; r < 8; ++r) {
      float a0 = s0[r] * scale + vol;
      float a1 = s1[r] * scale + vol;
      s0[r] = a0; s1[r] = a1;
      float t = fmaxf(a0, a1);
      t = fmaxf(t, __shfl_xor(t, 1, 32));
      t = fmaxf(t, __shfl_xor(t, 2, 32));
      t = fmaxf(t, __shfl_xor(t, 4, 32));
      t = fmaxf(t, __shfl_xor(t, 8, 32));
      float mn = fmaxf(mrun[r], t);
      alpha[r] = __expf(mrun[r] - mn);
      mnew[r]  = mn;
    }
#pragma unroll
    for (int r = 0; r < 8; ++r) {
      float p0 = __expf(s0[r] - mnew[r]);
      float p1 = __expf(s1[r] - mnew[r]);
      s0[r] = p0; s1[r] = p1;
      float t = p0 + p1;
      t += __shfl_xor(t, 1, 32);
      t += __shfl_xor(t, 2, 32);
      t += __shfl_xor(t, 4, 32);
      t += __shfl_xor(t, 8, 32);
      lrun[r] = lrun[r] * alpha[r] + t;
      mrun[r] = mnew[r];
    }
#pragma unroll
    for (int r = 0; r < 8; ++r) {
      o0[r] *= alpha[r]; o1[r] *= alpha[r]; o2[r] *= alpha[r]; o3[r] *= alpha[r];
    }

    // --- P: C-frag -> LDS (row-major 16x32 bf16) -> A-frag ---
#pragma unroll
    for (int r = 0; r < 8; ++r) {
      pw[(r + half * 8) * 32 + col]      = (__bf16)s0[r];
      pw[(r + half * 8) * 32 + col + 16] = (__bf16)s1[r];
    }
    v16bf pa = load_afrag(pw + col * 32, akoff);

    // --- O += P x V  (B-frag = contiguous rows of Vt) ---
    const __bf16* vrow = Vt + ((size_t)bh * DH_ + col) * L_ + j0 + bkoff;
    o0 = WMMA_BF16(pa, load_bfrag(vrow),             o0);
    o1 = WMMA_BF16(pa, load_bfrag(vrow + 16 * L_),   o1);
    o2 = WMMA_BF16(pa, load_bfrag(vrow + 32 * L_),   o2);
    o3 = WMMA_BF16(pa, load_bfrag(vrow + 48 * L_),   o3);
  }

  // --- normalize, write ctx (bf16, (B*L, D) row-major, head-interleaved) ---
#pragma unroll
  for (int r = 0; r < 8; ++r) {
    float inv = 1.0f / lrun[r];
    int lq = m0 + r + half * 8;
    size_t base = ((size_t)b * L_ + lq) * D_ + h * DH_;
    ctx[base + col]      = (__bf16)(o0[r] * inv);
    ctx[base + 16 + col] = (__bf16)(o1[r] * inv);
    ctx[base + 32 + col] = (__bf16)(o2[r] * inv);
    ctx[base + 48 + col] = (__bf16)(o3[r] * inv);
  }
}

// ---------------------------------------------------------------------------
// Kernel 4: out-proj GEMM + bias + residual + LayerNorm, fused per 16-row tile.
// 8 waves; wave w covers N = [w*128, w*128+128). Stage attended in LDS (64KB).
// ---------------------------------------------------------------------------
__global__ __launch_bounds__(256, 1)
void k_outln(const __bf16* __restrict__ ctx, const __bf16* __restrict__ wo,
             const float* __restrict__ bias, const float* __restrict__ x,
             const float* __restrict__ gamma, const float* __restrict__ beta,
             float* __restrict__ out) {
  __shared__ float att[16 * 1024];   // 64 KB

  const int lane  = threadIdx.x & 31;
  const int w     = threadIdx.x >> 5;
  const int m0    = blockIdx.x * 16;
  const int col   = lane & 15;
  const int half  = lane >> 4;
  const int akoff = half ? 16 : 0;
  const int bkoff = half ? 16 : 0;

  v8f zero = {0.f, 0.f, 0.f, 0.f, 0.f, 0.f, 0.f, 0.f};
  v8f c[8];
#pragma unroll
  for (int nt = 0; nt < 8; ++nt) c[nt] = zero;

  const __bf16* arow = ctx + (size_t)(m0 + col) * D_;
  for (int k0 = 0; k0 < D_; k0 += 32) {
    v16bf a = load_afrag(arow + k0, akoff);
#pragma unroll
    for (int nt = 0; nt < 8; ++nt) {
      const int n0 = w * 128 + nt * 16;
      const __bf16* brow = wo + (size_t)(n0 + col) * D_ + k0 + bkoff;
      c[nt] = WMMA_BF16(a, load_bfrag(brow), c[nt]);
    }
  }

#pragma unroll
  for (int nt = 0; nt < 8; ++nt) {
    const int e = w * 128 + nt * 16 + col;
    const float bv = bias[e];
#pragma unroll
    for (int r = 0; r < 8; ++r)
      att[(r + half * 8) * 1024 + e] = c[nt][r] + bv;
  }
  __syncthreads();

  // LayerNorm: 16 threads per row (consecutive tids = half-wave)
  const int row = threadIdx.x >> 4;
  const int sub = threadIdx.x & 15;
  const float* xrow = x + (size_t)(m0 + row) * D_;
  float sum = 0.f, ss = 0.f;
  for (int ci = sub; ci < D_; ci += 16) {
    float v = xrow[ci] + att[row * 1024 + ci];
    sum += v; ss += v * v;
  }
  sum += __shfl_xor(sum, 1, 32); sum += __shfl_xor(sum, 2, 32);
  sum += __shfl_xor(sum, 4, 32); sum += __shfl_xor(sum, 8, 32);
  ss  += __shfl_xor(ss, 1, 32);  ss  += __shfl_xor(ss, 2, 32);
  ss  += __shfl_xor(ss, 4, 32);  ss  += __shfl_xor(ss, 8, 32);
  float mean = sum * (1.f / 1024.f);
  float var  = ss * (1.f / 1024.f) - mean * mean;
  float rstd = rsqrtf(var + 1e-5f);

  float* orow = out + (size_t)(m0 + row) * D_;
  for (int ci = sub; ci < D_; ci += 16) {
    float v = xrow[ci] + att[row * 1024 + ci];
    orow[ci] = (v - mean) * rstd * gamma[ci] + beta[ci];
  }
}

// ---------------------------------------------------------------------------
extern "C" void kernel_launch(void* const* d_in, const int* in_sizes, int n_in,
                              void* d_out, int out_size, void* d_ws, size_t ws_size,
                              hipStream_t stream) {
  const float* x      = (const float*)d_in[0];
  const float* market = (const float*)d_in[1];
  const float* w_in   = (const float*)d_in[2];
  const float* b_in   = (const float*)d_in[3];
  const float* w_out  = (const float*)d_in[4];
  const float* b_out  = (const float*)d_in[5];
  const float* gamma  = (const float*)d_in[6];
  const float* beta   = (const float*)d_in[7];
  float* out = (float*)d_out;

  char* ws = (char*)d_ws;
  __bf16* xbf  = (__bf16*)(ws + 0);                         // 8 MB
  __bf16* wbf  = (__bf16*)(ws + 8388608);                   // 6 MB
  __bf16* wobf = (__bf16*)(ws + 14680064);                  // 2 MB
  __bf16* Qb   = (__bf16*)(ws + 16777216);                  // 8 MB
  __bf16* Kb   = (__bf16*)(ws + 25165824);                  // 8 MB
  __bf16* Vt   = (__bf16*)(ws + 33554432);                  // 8 MB
  __bf16* ctxb = (__bf16*)(ws + 41943040);                  // 8 MB

  k_f32_to_bf16<<<2048, 256, 0, stream>>>(x,     xbf,  BL_ * D_);
  k_f32_to_bf16<<<2048, 256, 0, stream>>>(w_in,  wbf,  D3_ * D_);
  k_f32_to_bf16<<<1024, 256, 0, stream>>>(w_out, wobf, D_ * D_);

  k_qkv<<<1536, 256, 0, stream>>>(xbf, wbf, b_in, Qb, Kb, Vt);
  k_flash<<<1024, 128, 0, stream>>>(Qb, Kb, Vt, market, ctxb);
  k_outln<<<256, 256, 0, stream>>>(ctxb, wobf, b_out, x, gamma, beta, out);
}
